// SpacetimeAttention_56873956934474
// MI455X (gfx1250) — matrix operand order repaired and checked by reading, not compiled, not run on hardware
//
#include <hip/hip_runtime.h>
#include <hip/hip_bf16.h>
#include <hip/hip_fp16.h>
#include <cstdint>

// Divided space-time attention for gfx1250 (MI455X).
// - All matrix math via v_wmma_f32_16x16x32_f16 (f32 accumulate).
// - GEMM staging via GLOBAL_LOAD_ASYNC_TO_LDS_B128 (ASYNCcnt-tracked DMA),
//   double-buffered LDS, one barrier per K-step.
// - V written pre-transposed by its projection GEMM so attention B-fragments
//   are contiguous b128 loads.

typedef __attribute__((ext_vector_type(16))) _Float16 v16h;
typedef __attribute__((ext_vector_type(8)))  _Float16 v8h;
typedef __attribute__((ext_vector_type(4)))  _Float16 v4h;
typedef __attribute__((ext_vector_type(8)))  float    v8f;
typedef __attribute__((ext_vector_type(4)))  float    v4f;

union F16x16 { v16h v; v8h h8[2]; };

#define D_EMB 384
#define NHEAD 6
#define DH    64
#define B_SZ  2
#define T_SZ  32
#define S_SZ  784
#define NTOK  (B_SZ * T_SZ * S_SZ)   // 50176 tokens

__device__ __forceinline__ v8f wmma_16x16x32(v16h a, v16h b, v8f c) {
  return __builtin_amdgcn_wmma_f32_16x16x32_f16(false, a, false, b, (short)0, c,
                                                false, false);
}

// A-fragment (16x32 f16): half j of lane L holds K = j + (L>=16)*16,
// i.e. one contiguous 16-half run at [hi*16].
__device__ __forceinline__ v16h loadA(const _Float16* row, int hi) {
  F16x16 a;
  a.h8[0] = *(const v8h*)(row + hi * 16);
  a.h8[1] = *(const v8h*)(row + hi * 16 + 8);
  return a.v;
}

// Async 16-byte global -> LDS copy (no VGPR round trip; ASYNCcnt-tracked).
__device__ __forceinline__ void async_b128(void* lds, const void* g) {
  uint32_t l = (uint32_t)(uintptr_t)lds;   // low 32 bits = LDS offset
  asm volatile("global_load_async_to_lds_b128 %0, %1, off"
               :: "v"(l), "v"((unsigned long long)(uintptr_t)g)
               : "memory");
}
__device__ __forceinline__ void wait_async0() {
  asm volatile("s_wait_asynccnt 0" ::: "memory");
}

// ---------------------------------------------------------------------------
// Elementwise f32 -> f16 conversion (for x / h feeding the QKV GEMMs).
// ---------------------------------------------------------------------------
__global__ __launch_bounds__(256)
void f32_to_f16_kernel(const float* __restrict__ in, _Float16* __restrict__ out,
                       int n) {
  int i = (blockIdx.x * 256 + threadIdx.x) * 4;
  if (i + 3 < n) {
    v4f f = *(const v4f*)&in[i];
    v4h h;
#pragma unroll
    for (int q = 0; q < 4; ++q) h[q] = (_Float16)f[q];
    *(v4h*)&out[i] = h;
  }
}

// ---------------------------------------------------------------------------
// Weight prep: W[K,N] f32 -> Wt[N,K] f16 (tiny, one-off).
// ---------------------------------------------------------------------------
__global__ __launch_bounds__(256)
void prep_weight_kernel(const float* __restrict__ W, _Float16* __restrict__ Wt,
                        int Kd, int Nd) {
  int idx = blockIdx.x * 256 + threadIdx.x;
  if (idx >= Kd * Nd) return;
  int k = idx % Kd, n = idx / Kd;          // writes contiguous along k
  Wt[(size_t)n * Kd + k] = (_Float16)W[(size_t)k * Nd + n];
}

// ---------------------------------------------------------------------------
// GEMM: out[M,N] = A[M,K](f16) @ Wt[N,K]^T(f16) + bias (+ residual f32).
// Block = 128 threads (4 waves), tile 64M x 128N, K-step 32.
// Double-buffered LDS fed by async-to-LDS b128 copies.
// vMode: 0 normal [M,N] f16 store; 1 spatial V^T; 2 temporal V^T.
// ---------------------------------------------------------------------------
__global__ __launch_bounds__(128)
void gemm_wt_kernel(const _Float16* __restrict__ A,
                    const _Float16* __restrict__ Wt,
                    const float* __restrict__ bias,
                    const float* __restrict__ residual,
                    float* __restrict__ outF32,
                    _Float16* __restrict__ outF16,
                    int vMode, int M, int Kd, int Nd) {
  __shared__ alignas(16) _Float16 aT[2][64 * 32];    // [row][k]
  __shared__ alignas(16) _Float16 wT[2][128 * 32];   // [n][k]

  const int lane = threadIdx.x & 31;
  const int wave = threadIdx.x >> 5;
  const int hi   = (lane >= 16) ? 1 : 0;
  const int ln   = lane & 15;
  const int mBase = blockIdx.x * 64;
  const int nBase = blockIdx.y * 128;
  const int nIter = Kd / 32;

  v8f acc[8];
#pragma unroll
  for (int t = 0; t < 8; ++t) acc[t] = (v8f){0.f,0.f,0.f,0.f,0.f,0.f,0.f,0.f};

  // Stage tile (k0) into buffer buf: pure b128 async copies.
  auto stage = [&](int buf, int k0) {
#pragma unroll
    for (int s = 0; s < 2; ++s) {            // A: 256 segs of 16B
      int seg = threadIdx.x * 2 + s;
      int row = seg >> 2, part = seg & 3;
      async_b128(&aT[buf][row * 32 + part * 8],
                 A + (size_t)(mBase + row) * Kd + k0 + part * 8);
    }
#pragma unroll
    for (int s = 0; s < 4; ++s) {            // Wt: 512 segs of 16B
      int seg = threadIdx.x * 4 + s;
      int row = seg >> 2, part = seg & 3;
      async_b128(&wT[buf][row * 32 + part * 8],
                 Wt + (size_t)(nBase + row) * Kd + k0 + part * 8);
    }
  };

  stage(0, 0);
  int buf = 0;
  for (int it = 0; it < nIter; ++it) {
    wait_async0();            // current tile's copies landed in LDS
    __syncthreads();          // visible to all waves
    if (it + 1 < nIter) stage(buf ^ 1, (it + 1) * 32);  // overlap next copies

    v16h aF = loadA(&aT[buf][(wave * 16 + ln) * 32], hi);
#pragma unroll
    for (int t = 0; t < 8; ++t) {
      // B-fragment: half j holds K = j + hi*16 -> contiguous 16 halves.
      v16h bF = *(const v16h*)&wT[buf][(t * 16 + ln) * 32 + hi * 16];
      acc[t] = wmma_16x16x32(aF, bF, acc[t]);
    }
    buf ^= 1;
  }

  // Epilogue: C layout lane holds N = ln, VGPR r holds row r + hi*8.
#pragma unroll
  for (int t = 0; t < 8; ++t) {
    int col = nBase + t * 16 + ln;
    float bv = bias ? bias[col] : 0.f;
#pragma unroll
    for (int r = 0; r < 8; ++r) {
      int row = mBase + wave * 16 + r + hi * 8;
      if (row >= M) continue;
      size_t off = (size_t)row * Nd + col;
      float v = acc[t][r] + bv;
      if (residual) v += residual[off];
      if (outF32) outF32[off] = v;
      if (outF16) {
        if (vMode == 0) {
          outF16[off] = (_Float16)v;
        } else {
          int h = col / DH, d = col % DH;
          size_t toff;
          if (vMode == 1) {                   // spatial: seq=(b*T+t), key=s
            int seq = row / S_SZ, s = row % S_SZ;
            toff = ((size_t)seq * NHEAD + h) * ((size_t)DH * S_SZ)
                 + (size_t)d * S_SZ + s;
          } else {                            // temporal: seq=(b*S+s), key=t
            int b  = row / (T_SZ * S_SZ);
            int rm = row % (T_SZ * S_SZ);
            int tt = rm / S_SZ, s = rm % S_SZ;
            int seq = b * S_SZ + s;
            toff = ((size_t)seq * NHEAD + h) * ((size_t)DH * T_SZ)
                 + (size_t)d * T_SZ + tt;
          }
          outF16[toff] = (_Float16)v;
        }
      }
    }
  }
}

// ---------------------------------------------------------------------------
// Flash attention over one axis. 4 waves/block; each wave owns one
// 16-query tile of one (sequence, head). Q/K in [B,T,S,H,dh]; V transposed
// per (seq, head): vT[seq][h][d][key].
// ---------------------------------------------------------------------------
__global__ __launch_bounds__(128)
void flash_attn_kernel(const _Float16* __restrict__ Q,
                       const _Float16* __restrict__ K,
                       const _Float16* __restrict__ vT,
                       _Float16* __restrict__ O,
                       int spatial, int Lseq, int numQT, int totalTiles) {
  __shared__ alignas(16) _Float16 pbuf[4][16 * 32];

  const int lane = threadIdx.x & 31;
  const int wave = threadIdx.x >> 5;
  const int hi   = (lane >= 16) ? 1 : 0;
  const int ln   = lane & 15;

  int tile = blockIdx.x * 4 + wave;
  if (tile >= totalTiles) return;
  int qt   = tile % numQT;
  int h    = (tile / numQT) % NHEAD;
  int sIdx = tile / (numQT * NHEAD);

  size_t base; long rowStride;
  if (spatial) {                        // seq axis = S, one seq per (b,t)
    base = (size_t)sIdx * S_SZ * D_EMB + (size_t)h * DH;
    rowStride = D_EMB;
  } else {                              // seq axis = T, one seq per (b,s)
    int b = sIdx / S_SZ, s = sIdx % S_SZ;
    base = (size_t)b * T_SZ * S_SZ * D_EMB + (size_t)s * D_EMB + (size_t)h * DH;
    rowStride = (long)S_SZ * D_EMB;
  }
  const size_t vBase = ((size_t)sIdx * NHEAD + h) * ((size_t)DH * Lseq);
  _Float16* pb = pbuf[wave];

  const _Float16* qrow = Q + base + (size_t)(qt * 16 + ln) * rowStride;
  v16h aQ0 = loadA(qrow, hi);
  v16h aQ1 = loadA(qrow + 32, hi);

  v8f accO[4];
#pragma unroll
  for (int t = 0; t < 4; ++t) accO[t] = (v8f){0.f,0.f,0.f,0.f,0.f,0.f,0.f,0.f};
  float mr[8], lr[8];
#pragma unroll
  for (int r = 0; r < 8; ++r) { mr[r] = -1e30f; lr[r] = 0.f; }
  const float scale = 0.125f;           // 1/sqrt(64)

  for (int k0 = 0; k0 < Lseq; k0 += 32) {
    // ---- scores: two 16x16 tiles -------------------------------------
    v8f sc[2];
#pragma unroll
    for (int t = 0; t < 2; ++t) {
      int key = k0 + t * 16 + ln;
      v16h bLo, bHi;                    // B: half j holds dim d = j + hi*16
      if (key < Lseq) {
        const _Float16* krow = K + base + (size_t)key * rowStride;
        bLo = *(const v16h*)(krow + hi * 16);
        bHi = *(const v16h*)(krow + 32 + hi * 16);
        if (key + 32 < Lseq)
          __builtin_prefetch(krow + 32 * rowStride, 0, 1);
      } else {
#pragma unroll
        for (int j = 0; j < 16; ++j) { bLo[j] = (_Float16)0.f; bHi[j] = (_Float16)0.f; }
      }
      v8f c = (v8f){0.f,0.f,0.f,0.f,0.f,0.f,0.f,0.f};
      c = wmma_16x16x32(aQ0, bLo, c);
      c = wmma_16x16x32(aQ1, bHi, c);
#pragma unroll
      for (int r = 0; r < 8; ++r)
        sc[t][r] = (key < Lseq) ? c[r] * scale : -1e30f;
    }

    // ---- online softmax (rows live in VGPR r + half-group hi) --------
    float p0[8], p1[8];
#pragma unroll
    for (int r = 0; r < 8; ++r) {
      float tm = fmaxf(sc[0][r], sc[1][r]);
#pragma unroll
      for (int off = 1; off < 16; off <<= 1)
        tm = fmaxf(tm, __shfl_xor(tm, off, 32));
      float mn = fmaxf(mr[r], tm);
      float alpha = __expf(mr[r] - mn);
      mr[r] = mn;
      p0[r] = __expf(sc[0][r] - mn);
      p1[r] = __expf(sc[1][r] - mn);
      float ps = p0[r] + p1[r];
#pragma unroll
      for (int off = 1; off < 16; off <<= 1)
        ps += __shfl_xor(ps, off, 32);
      lr[r] = lr[r] * alpha + ps;
#pragma unroll
      for (int t = 0; t < 4; ++t) accO[t][r] *= alpha;
    }

    // ---- P (16x32) through LDS into A-layout -------------------------
#pragma unroll
    for (int r = 0; r < 8; ++r) {
      pb[(r + hi * 8) * 32 + ln]      = (_Float16)p0[r];
      pb[(r + hi * 8) * 32 + 16 + ln] = (_Float16)p1[r];
    }
    asm volatile("s_wait_dscnt 0" ::: "memory");
    v16h aP = loadA(&pb[ln * 32], hi);

    // ---- O += P @ V : contiguous B-fragments from transposed V -------
#pragma unroll
    for (int t = 0; t < 4; ++t) {
      const _Float16* vrow = vT + vBase + (size_t)(t * 16 + ln) * Lseq;
      v16h bV;
      if (k0 + hi * 16 < Lseq) {        // 16-half group fully valid or not
        bV = *(const v16h*)(vrow + k0 + hi * 16);
        if (k0 + 32 + hi * 16 < Lseq)
          __builtin_prefetch(vrow + k0 + 32 + hi * 16, 0, 1);
      } else {
#pragma unroll
        for (int j = 0; j < 16; ++j) bV[j] = (_Float16)0.f;
      }
      accO[t] = wmma_16x16x32(aP, bV, accO[t]);
    }
  }

  // ---- normalize and store O (f16, [token,D] layout) -----------------
#pragma unroll
  for (int t = 0; t < 4; ++t) {
#pragma unroll
    for (int r = 0; r < 8; ++r) {
      float v = accO[t][r] / lr[r];
      size_t orow = base + (size_t)(qt * 16 + r + hi * 8) * rowStride;
      O[orow + t * 16 + ln] = (_Float16)v;
    }
  }
}

// ---------------------------------------------------------------------------
// Host-side launch.
// ---------------------------------------------------------------------------
extern "C" void kernel_launch(void* const* d_in, const int* in_sizes, int n_in,
                              void* d_out, int out_size, void* d_ws, size_t ws_size,
                              hipStream_t stream) {
  const float* x = (const float*)d_in[0];
  const float* Wmat[8] = { (const float*)d_in[1],  (const float*)d_in[3],
                           (const float*)d_in[5],  (const float*)d_in[7],
                           (const float*)d_in[9],  (const float*)d_in[11],
                           (const float*)d_in[13], (const float*)d_in[15] };
  const float* bvec[8] = { (const float*)d_in[2],  (const float*)d_in[4],
                           (const float*)d_in[6],  (const float*)d_in[8],
                           (const float*)d_in[10], (const float*)d_in[12],
                           (const float*)d_in[14], (const float*)d_in[16] };

  char* ws = (char*)d_ws;
  const size_t halfMat = (size_t)NTOK * D_EMB * sizeof(_Float16); // 38.5 MB
  const size_t wtBytes = (size_t)D_EMB * D_EMB * sizeof(_Float16);
  _Float16* qb  = (_Float16*)(ws);
  _Float16* kb  = (_Float16*)(ws + halfMat);
  _Float16* vtb = (_Float16*)(ws + 2 * halfMat);   // transposed V
  _Float16* ob  = (_Float16*)(ws + 3 * halfMat);
  _Float16* fh  = (_Float16*)(ws + 4 * halfMat);   // f16 copy of x, then h
  float*    hb  = (float*)   (ws + 5 * halfMat);   // f32 hidden (77 MB)
  _Float16* wt[8];
  {
    char* p = ws + 5 * halfMat + (size_t)NTOK * D_EMB * sizeof(float);
    for (int i = 0; i < 8; ++i) wt[i] = (_Float16*)(p + i * wtBytes);
  }

  const int nElem = NTOK * D_EMB;
  f32_to_f16_kernel<<<dim3(nElem / (256 * 4)), dim3(256), 0, stream>>>(x, fh, nElem);
  for (int i = 0; i < 8; ++i)
    prep_weight_kernel<<<dim3((D_EMB * D_EMB + 255) / 256), dim3(256), 0, stream>>>(
        Wmat[i], wt[i], D_EMB, D_EMB);

  dim3 gG(NTOK / 64, D_EMB / 128);   // 784 x 3
  dim3 bG(128);

  // ---- spatial MHA ----
  gemm_wt_kernel<<<gG, bG, 0, stream>>>(fh, wt[0], bvec[0], nullptr, nullptr, qb,  0, NTOK, D_EMB, D_EMB);
  gemm_wt_kernel<<<gG, bG, 0, stream>>>(fh, wt[1], bvec[1], nullptr, nullptr, kb,  0, NTOK, D_EMB, D_EMB);
  gemm_wt_kernel<<<gG, bG, 0, stream>>>(fh, wt[2], bvec[2], nullptr, nullptr, vtb, 1, NTOK, D_EMB, D_EMB);

  int nQTs = S_SZ / 16;                         // 49
  int tilesS = B_SZ * T_SZ * NHEAD * nQTs;      // 18816
  flash_attn_kernel<<<dim3(tilesS / 4), dim3(128), 0, stream>>>(
      qb, kb, vtb, ob, /*spatial=*/1, S_SZ, nQTs, tilesS);

  // h = x + o @ Wo_s + bo_s   (f32 into hb, f16 copy into fh)
  gemm_wt_kernel<<<gG, bG, 0, stream>>>(ob, wt[3], bvec[3], x, hb, fh, 0, NTOK, D_EMB, D_EMB);

  // ---- temporal MHA ----
  gemm_wt_kernel<<<gG, bG, 0, stream>>>(fh, wt[4], bvec[4], nullptr, nullptr, qb,  0, NTOK, D_EMB, D_EMB);
  gemm_wt_kernel<<<gG, bG, 0, stream>>>(fh, wt[5], bvec[5], nullptr, nullptr, kb,  0, NTOK, D_EMB, D_EMB);
  gemm_wt_kernel<<<gG, bG, 0, stream>>>(fh, wt[6], bvec[6], nullptr, nullptr, vtb, 2, NTOK, D_EMB, D_EMB);

  int nQTt = T_SZ / 16;                         // 2
  int tilesT = B_SZ * S_SZ * NHEAD * nQTt;      // 18816
  flash_attn_kernel<<<dim3(tilesT / 4), dim3(128), 0, stream>>>(
      qb, kb, vtb, ob, /*spatial=*/0, T_SZ, nQTt, tilesT);

  // out = h + o @ Wo_t + bo_t
  gemm_wt_kernel<<<gG, bG, 0, stream>>>(ob, wt[7], bvec[7], hb, (float*)d_out, nullptr, 0, NTOK, D_EMB, D_EMB);
}
